// Model_70824010711618
// MI455X (gfx1250) — matrix operand order repaired and checked by reading, not compiled, not run on hardware
//
#include <hip/hip_runtime.h>

#define RNN   128
#define GATES 512
#define MT    2                 // two 16-row M tiles per wave (32 batch rows)
#define WAVES 8
#define ROWS_PER_WG (WAVES * MT * 16)   // 256

typedef __attribute__((ext_vector_type(16))) __bf16 v16bf;
typedef __attribute__((ext_vector_type(8)))  __bf16 v8bf;
typedef __attribute__((ext_vector_type(8)))  float  v8f;
typedef __attribute__((ext_vector_type(8)))  unsigned short v8us;

__device__ __forceinline__ unsigned short f2bf(float f) {
  unsigned int u = __builtin_bit_cast(unsigned int, f);
  u += 0x7FFFu + ((u >> 16) & 1u);   // round-to-nearest-even
  return (unsigned short)(u >> 16);
}
__device__ __forceinline__ float bf2f(unsigned short h) {
  unsigned int u = ((unsigned int)h) << 16;
  return __builtin_bit_cast(float, u);
}
__device__ __forceinline__ float tanh_hw(float x) {
#if __has_builtin(__builtin_amdgcn_tanhf)
  return __builtin_amdgcn_tanhf(x);        // native V_TANH_F32 (TRANS pipe)
#else
  float e = __expf(2.0f * x);
  return 1.0f - 2.0f / (e + 1.0f);
#endif
}
__device__ __forceinline__ float sigmoid_hw(float x) {
  return __builtin_fmaf(0.5f, tanh_hw(0.5f * x), 0.5f);  // sig(x)=0.5*tanh(x/2)+0.5
}

// LSTM trajectory predictor: 19 recurrent steps, h@W_hh^T via bf16 WMMA (f32 acc).
// Each wave owns 32 batch rows (2 WMMA M-tiles) so every B fragment feeds 2 WMMAs.
__global__ __launch_bounds__(256, 1)
void lstm_traj_kernel(const float* __restrict__ pose,   // (B,3,10,18,1)
                      const float* __restrict__ Wih,    // (512,2)
                      const float* __restrict__ Whh,    // (512,128)
                      const float* __restrict__ bih,    // (512)
                      const float* __restrict__ bhh,    // (512)
                      const float* __restrict__ Wlast,  // (2,128)
                      const float* __restrict__ blast,  // (2)
                      float* __restrict__ out,          // (B,10,2)
                      int Btot) {
  // B-matrix: W_hh^T column-major == W_hh row-major [n][k], bf16
  __shared__ __align__(16) unsigned short sh_whh[GATES * RNN];        // 128 KB
  __shared__ __align__(16) unsigned short sh_h[WAVES][MT * 16][RNN];  // 64 KB
  __shared__ float sh_x[ROWS_PER_WG][10][2];                          // 20 KB
  __shared__ float sh_gb[GATES];                                      // b_ih + b_hh
  __shared__ float sh_wi0[GATES];
  __shared__ float sh_wi1[GATES];
  __shared__ float sh_wl[2 * RNN];
  __shared__ float sh_bl[2];
  __shared__ float sh_p[ROWS_PER_WG][2];                              // feedback input

  const int tid = threadIdx.x;
  const long rowbase = (long)blockIdx.x * ROWS_PER_WG;

  // ---------------- staging (once per workgroup) ----------------
  for (int i = tid; i < GATES * RNN; i += 256) sh_whh[i] = f2bf(Whh[i]);
  for (int n = tid; n < GATES; n += 256) {
    sh_gb[n]  = bih[n] + bhh[n];
    sh_wi0[n] = Wih[2 * n + 0];
    sh_wi1[n] = Wih[2 * n + 1];
  }
  if (tid < 2 * RNN) sh_wl[tid] = Wlast[tid];
  if (tid < 2)       sh_bl[tid] = blast[tid];
  {
    unsigned short* hz = &sh_h[0][0][0];
    for (int i = tid; i < WAVES * MT * 16 * RNN; i += 256) hz[i] = 0; // h0 = 0
  }
  // mid-hip input: x[b,t,d] = 0.5*(pose[b,d,t,8] + pose[b,d,t,11])
  for (int idx = tid; idx < ROWS_PER_WG * 10 * 2; idx += 256) {
    int d = idx & 1;
    int t = (idx >> 1) % 10;
    int r = idx / 20;
    long b = rowbase + r;
    float v = 0.0f;
    if (b < (long)Btot) {
      long base = ((b * 3 + d) * 10 + t) * 18;
      v = 0.5f * (pose[base + 8] + pose[base + 11]);
    }
    sh_x[r][t][d] = v;
  }
  __syncthreads();

  const int lane  = tid & 31;
  const int w     = tid >> 5;        // wave id: owns batch rows [w*32, w*32+32)
  const int nlane = lane & 15;       // C/D layout: N column; A layout: M row
  const int khalf = lane >> 4;
  const int mgrp  = khalf * 8;       // C/D layout: M rows mgrp..mgrp+7

  // cell state: creg[mt][j*8+r] = c[mt*16 + mgrp+r][j*16+nlane]
  float creg[MT][64];
  #pragma unroll
  for (int t = 0; t < MT; ++t)
    #pragma unroll
    for (int i = 0; i < 64; ++i) creg[t][i] = 0.0f;

  #pragma unroll 1
  for (int step = 0; step < 19; ++step) {
    // per-row input (x_t for obs steps, fed-back prediction for pred steps)
    float x0r[MT][8], x1r[MT][8];
    #pragma unroll
    for (int t = 0; t < MT; ++t) {
      #pragma unroll
      for (int r = 0; r < 8; ++r) {
        int row = w * (MT * 16) + t * 16 + mgrp + r;
        if (step < 10) { x0r[t][r] = sh_x[row][step][0]; x1r[t][r] = sh_x[row][step][1]; }
        else           { x0r[t][r] = sh_p[row][0];       x1r[t][r] = sh_p[row][1];       }
      }
    }

    // A fragments: h tiles (16 x 128 each), ISA A layout:
    // lane holds row M=nlane; K = khalf*8 + j (j<8), 16 + khalf*8 + (j-8) (j>=8)
    v16bf af[MT][4];
    #pragma unroll
    for (int t = 0; t < MT; ++t) {
      #pragma unroll
      for (int kt = 0; kt < 4; ++kt) {
        const unsigned short* ab = &sh_h[w][t * 16 + nlane][kt * 32 + khalf * 8];
        v8bf lo = *(const v8bf*)ab;
        v8bf hi = *(const v8bf*)(ab + 16);
        af[t][kt] = __builtin_shufflevector(lo, hi, 0,1,2,3,4,5,6,7,8,9,10,11,12,13,14,15);
      }
    }

    #pragma unroll
    for (int j = 0; j < 8; ++j) {
      const int n0 = j * 16 + nlane;
      // C init: bias + x @ W_ih^T for the four gates (i,f,g,o), both M tiles
      v8f acc[MT][4];
      #pragma unroll
      for (int g = 0; g < 4; ++g) {
        const int n = g * RNN + n0;
        float gb = sh_gb[n], w0 = sh_wi0[n], w1 = sh_wi1[n];
        #pragma unroll
        for (int t = 0; t < MT; ++t)
          #pragma unroll
          for (int r = 0; r < 8; ++r)
            acc[t][g][r] = gb + w0 * x0r[t][r] + w1 * x1r[t][r];
      }
      // gates += h @ W_hh^T : each B fragment feeds both M tiles
      #pragma unroll
      for (int g = 0; g < 4; ++g) {
        const int n = g * RNN + n0;
        #pragma unroll
        for (int kt = 0; kt < 4; ++kt) {
          // B layout: lane holds column N=nlane; K = khalf*8 + j (j<8), 16 + khalf*8 + (j-8) (j>=8)
          const unsigned short* bb = &sh_whh[n * RNN + kt * 32 + khalf * 8];
          v8bf lo = *(const v8bf*)bb;
          v8bf hi = *(const v8bf*)(bb + 16);
          v16bf bfr = __builtin_shufflevector(lo, hi, 0,1,2,3,4,5,6,7,8,9,10,11,12,13,14,15);
          #pragma unroll
          for (int t = 0; t < MT; ++t)
            acc[t][g] = __builtin_amdgcn_wmma_f32_16x16x32_bf16(
                false, af[t][kt], false, bfr, (short)0, acc[t][g], false, false);
        }
      }
      // activations + cell/hidden update (C/D register layout, no shuffles)
      #pragma unroll
      for (int t = 0; t < MT; ++t) {
        #pragma unroll
        for (int r = 0; r < 8; ++r) {
          float ig = sigmoid_hw(acc[t][0][r]);
          float fg = sigmoid_hw(acc[t][1][r]);
          float gg = tanh_hw(acc[t][2][r]);
          float og = sigmoid_hw(acc[t][3][r]);
          float c  = fg * creg[t][j * 8 + r] + ig * gg;
          creg[t][j * 8 + r] = c;
          float h = og * tanh_hw(c);
          sh_h[w][t * 16 + mgrp + r][n0] = f2bf(h);
        }
      }
    }

    // head: relu(h) @ W_last^T + b_last, needed after steps 9..18 (outputs k=0..9)
    if (step >= 9) {
      #pragma unroll
      for (int t = 0; t < MT; ++t) {
        const int row32 = t * 16 + nlane;   // 32 rows x 2 outputs over 32 lanes x 2 passes
        const int oi    = khalf;
        float s = sh_bl[oi];
        #pragma unroll
        for (int kb = 0; kb < 16; ++kb) {
          v8us ch = *(const v8us*)&sh_h[w][row32][kb * 8];
          #pragma unroll
          for (int e = 0; e < 8; ++e) {
            float hv = bf2f(ch[e]);
            s += fmaxf(hv, 0.0f) * sh_wl[oi * RNN + kb * 8 + e];
          }
        }
        int  rloc = w * (MT * 16) + row32;
        long b    = rowbase + rloc;
        if (b < (long)Btot) out[b * 20 + (long)(step - 9) * 2 + oi] = s;
        sh_p[rloc][oi] = s;   // feedback input for next step
      }
    }
    __syncthreads();
  }
}

extern "C" void kernel_launch(void* const* d_in, const int* in_sizes, int n_in,
                              void* d_out, int out_size, void* d_ws, size_t ws_size,
                              hipStream_t stream) {
  // dict order: obs_loc, obs_pose, obs_gridflow, W_ih, W_hh, b_ih, b_hh, W_last, b_last
  const float* pose  = (const float*)d_in[1];
  const float* Wih   = (const float*)d_in[3];
  const float* Whh   = (const float*)d_in[4];
  const float* bih   = (const float*)d_in[5];
  const float* bhh   = (const float*)d_in[6];
  const float* Wlast = (const float*)d_in[7];
  const float* blast = (const float*)d_in[8];
  float* out = (float*)d_out;

  int B = in_sizes[1] / (3 * 10 * 18);
  int nwg = (B + ROWS_PER_WG - 1) / ROWS_PER_WG;
  lstm_traj_kernel<<<dim3(nwg), dim3(256), 0, stream>>>(
      pose, Wih, Whh, bih, bhh, Wlast, blast, out, B);
}
